// GRINCell_90915867722322
// MI455X (gfx1250) — matrix operand order repaired and checked by reading, not compiled, not run on hardware
//
#include <hip/hip_runtime.h>
#include <hip/hip_bf16.h>
#include <math.h>

// ---------------- problem constants (match reference) ----------------
#define NN 3000   // nodes
#define BB 8      // batch
#define TT 24     // time steps
#define EE 48000  // edges
#define HH 64     // hidden
#define RR (BB * NN)  // 24000 rows, multiple of 16

typedef __attribute__((ext_vector_type(16))) _Float16 v16h;
typedef __attribute__((ext_vector_type(8)))  float    v8f;

// ---------------- weight pre-pack: W[K x 64] fp32 -> f16 B-fragments ----------------
// Fragment order: [kt][coltile(4)][lane(32)][16 halfs]; per-lane 16 halfs contiguous.
// B layout used here (contiguous k, same as A loads): N = lane&15 ; lanes<16 K = kt*32+{0..15}, lanes>=16 K = kt*32+{16..31}
__global__ void pack_b_kernel(const float* __restrict__ W, int K, int KPAD,
                              _Float16* __restrict__ out) {
  int idx = blockIdx.x * blockDim.x + threadIdx.x;
  int total = (KPAD / 32) * 4 * 32;
  if (idx >= total) return;
  int lane = idx & 31;
  int ct   = (idx >> 5) & 3;
  int kt   = idx >> 7;
  int n  = lane & 15;
  int kb = (lane >> 4) << 4;  // 0 or 16
  _Float16* o = out + (size_t)idx * 16;
#pragma unroll
  for (int i = 0; i < 16; ++i) {
    int k = kt * 32 + kb + i;
    o[i] = (k < K) ? (_Float16)W[(size_t)k * 64 + ct * 16 + n] : (_Float16)0.0f;
  }
}

// ---------------- WMMA matmul: C[RR x 64] = act(A[RR x KPAD] @ W + bias) ----------------
// A is fp32 with lda == KPAD (pad columns zero). Bp is the packed f16 fragment buffer.
// block = 128 threads = 4 wave32, one wave per 16-col tile; grid.x = RR/16.
// act: 0 = none, 1 = sigmoid, 2 = tanh, 3 = PReLU(alpha)
template <int KPAD>
__global__ void wmma_mm_kernel(const float* __restrict__ A,
                               const _Float16* __restrict__ Bp,
                               const float* __restrict__ bias,
                               float* __restrict__ C, int ldc,
                               int act, const float* __restrict__ alphap) {
  constexpr int KT = KPAD / 32;
  int wave = threadIdx.x >> 5;
  int lane = threadIdx.x & 31;
  int row0 = blockIdx.x * 16;
  int m  = lane & 15;
  int kh = (lane >> 4) << 4;  // A: lanes<16 K {0..15}; lanes>=16 {16..31}
  const float* rp = A + (size_t)(row0 + m) * KPAD + kh;
  const _Float16* bp = Bp + ((size_t)wave * 32 + lane) * 16;

  v8f acc = {};
#pragma unroll
  for (int kt = 0; kt < KT; ++kt) {
    const float4* p0 = (const float4*)(rp + kt * 32);       // K = kt*32 + kh + {0..7}
    const float4* p1 = (const float4*)(rp + kt * 32 + 8);   // K = kt*32 + kh + {8..15}
    float4 f0 = p0[0], f1 = p0[1], f2 = p1[0], f3 = p1[1];
    v16h a;
    a[0]  = (_Float16)f0.x; a[1]  = (_Float16)f0.y; a[2]  = (_Float16)f0.z; a[3]  = (_Float16)f0.w;
    a[4]  = (_Float16)f1.x; a[5]  = (_Float16)f1.y; a[6]  = (_Float16)f1.z; a[7]  = (_Float16)f1.w;
    a[8]  = (_Float16)f2.x; a[9]  = (_Float16)f2.y; a[10] = (_Float16)f2.z; a[11] = (_Float16)f2.w;
    a[12] = (_Float16)f3.x; a[13] = (_Float16)f3.y; a[14] = (_Float16)f3.z; a[15] = (_Float16)f3.w;
    v16h b = *(const v16h*)(bp + (size_t)kt * (4 * 32 * 16));
    acc = __builtin_amdgcn_wmma_f32_16x16x32_f16(
        /*neg_a=*/false, a, /*neg_b=*/false, b,
        /*c_mod=*/(short)0, acc, /*reuse_a=*/false, /*reuse_b=*/false);
  }

  int n  = lane & 15;
  int mb = (lane >> 4) * 8;  // C layout: VGPR i -> M = mb + i
  int col0 = wave * 16;
  float bv = bias ? bias[col0 + n] : 0.0f;
  float al = (act == 3 && alphap) ? alphap[0] : 0.0f;
#pragma unroll
  for (int i = 0; i < 8; ++i) {
    float v = acc[i] + bv;
    if (act == 1)      v = 1.0f / (1.0f + __expf(-v));
    else if (act == 2) v = tanhf(v);
    else if (act == 3) v = (v >= 0.0f) ? v : al * v;
    C[(size_t)(row0 + mb + i) * ldc + col0 + n] = v;
  }
}

// ---------------- graph normalization ----------------
__global__ void zero_f32_kernel(float* __restrict__ p, int n) {
  int i = blockIdx.x * blockDim.x + threadIdx.x;
  if (i < n) p[i] = 0.0f;
}

__global__ void deg_accum_kernel(const int* __restrict__ src, const int* __restrict__ dst,
                                 const float* __restrict__ w,
                                 float* __restrict__ degf, float* __restrict__ degb) {
  int e = blockIdx.x * blockDim.x + threadIdx.x;
  if (e >= EE) return;
  atomicAdd(&degf[dst[e]], w[e]);
  atomicAdd(&degb[src[e]], w[e]);
}

__global__ void norm_w_kernel(const int* __restrict__ src, const int* __restrict__ dst,
                              const float* __restrict__ w,
                              const float* __restrict__ degf, const float* __restrict__ degb,
                              float* __restrict__ wf, float* __restrict__ wb) {
  int e = blockIdx.x * blockDim.x + threadIdx.x;
  if (e >= EE) return;
  float df = degf[dst[e]];
  float db = degb[src[e]];
  wf[e] = (df > 0.0f) ? w[e] / df : 0.0f;
  wb[e] = (db > 0.0f) ? w[e] / db : 0.0f;
}

__global__ void init_h_kernel(const float* __restrict__ h0, float* __restrict__ h) {
  int i = blockIdx.x * blockDim.x + threadIdx.x;
  if (i >= RR * HH) return;
  int r = i / HH, c = i - r * HH;
  int n = r % NN;
  h[i] = h0[(size_t)n * HH + c];
}

// one directional diffusion hop: out[b,d,f] += w[e] * in[b,s,f]
__global__ void prop_dir_kernel(const float* __restrict__ in, int ldin,
                                float* __restrict__ out, int ldout, int F,
                                const int* __restrict__ sIdx, const int* __restrict__ dIdx,
                                const float* __restrict__ w) {
  int idx = blockIdx.x * blockDim.x + threadIdx.x;
  if (idx >= EE * F) return;
  int e = idx / F, f = idx - e * F;
  int s = sIdx[e], d = dIdx[e];
  float wv = w[e];
  if (wv == 0.0f) return;
#pragma unroll
  for (int b = 0; b < BB; ++b) {
    float v = in[(size_t)(b * NN + s) * ldin + f] * wv;
    atomicAdd(&out[(size_t)(b * NN + d) * ldout + f], v);
  }
}

// ---------------- fused stage kernels (one thread per (batch,node) row) ----------------
// pre: xhat1 = h@W_fs+b (-> predictions), x1 = sel(mask,x,xhat1),
//      buf96[:,0:72] = [x1|m|h], b128b[:,64:128]=h, b128c[:,64:128]=h, zero b128a
__global__ void pre_kernel(const float* __restrict__ x, const float* __restrict__ mask,
                           const float* __restrict__ h,
                           const float* __restrict__ Wfs, const float* __restrict__ bfs,
                           float* __restrict__ pred_out, float* __restrict__ buf96,
                           float* __restrict__ b128a, float* __restrict__ b128b,
                           float* __restrict__ b128c, int t) {
  int r = blockIdx.x * blockDim.x + threadIdx.x;
  if (r >= RR) return;
  int b = r / NN, n = r - b * NN;
  size_t io = ((size_t)(b * TT + t) * NN + n) * 4;
  const float* hr = h + (size_t)r * HH;
  float* z = buf96 + (size_t)r * 96;
#pragma unroll
  for (int f = 0; f < 4; ++f) {
    float acc = bfs[f];
    for (int k = 0; k < HH; ++k) acc += hr[k] * Wfs[k * 4 + f];
    pred_out[io + f] = acc;
    float m = mask[io + f];
    z[f] = (m > 0.5f) ? x[io + f] : acc;  // x1
    z[4 + f] = m;
  }
  float* a  = b128a + (size_t)r * 128;
  float* bB = b128b + (size_t)r * 128;
  float* cB = b128c + (size_t)r * 128;
  for (int c = 0; c < HH; ++c) {
    float hv = hr[c];
    z[8 + c]   = hv;
    bB[64 + c] = hv;
    cB[64 + c] = hv;
    a[c] = 0.0f;
    a[64 + c] = 0.0f;
  }
}

// mid: xhat2 = rep@W_ro+b (-> imputations), x2 = sel(mask,x1,xhat2),
//      representations = rep, b384[:,0:72] = [x2|m|h], zero b384[:,72:360]
__global__ void mid_kernel(const float* __restrict__ mask, const float* __restrict__ buf96,
                           const float* __restrict__ b128c,
                           const float* __restrict__ Wro, const float* __restrict__ bro,
                           float* __restrict__ imp_out, float* __restrict__ reps_out,
                           float* __restrict__ b384, int t) {
  int r = blockIdx.x * blockDim.x + threadIdx.x;
  if (r >= RR) return;
  int b = r / NN, n = r - b * NN;
  size_t io = ((size_t)(b * TT + t) * NN + n) * 4;
  const float* rep = b128c + (size_t)r * 128;
  float* g = b384 + (size_t)r * 384;
#pragma unroll
  for (int f = 0; f < 4; ++f) {
    float acc = bro[f];
    for (int k = 0; k < 128; ++k) acc += rep[k] * Wro[k * 4 + f];
    imp_out[io + f] = acc;  // imputations collect raw xhat2
    float m = mask[io + f];
    g[f] = (m > 0.5f) ? buf96[(size_t)r * 96 + f] : acc;  // x2
    g[4 + f] = m;
  }
  size_t ro = ((size_t)(b * TT + t) * NN + n) * 128;
  for (int c = 0; c < 128; ++c) reps_out[ro + c] = rep[c];
  for (int c = 0; c < HH; ++c) g[8 + c] = rep[64 + c];  // h (placed there by pre)
  for (int c = 72; c < 360; ++c) g[c] = 0.0f;
}

// candprep: b384[:,8:72] = r*h ; zero b384[:,72:360] (cols 0..7 keep xi)
__global__ void candprep_kernel(const float* __restrict__ rbuf, const float* __restrict__ h,
                                float* __restrict__ b384) {
  int r = blockIdx.x * blockDim.x + threadIdx.x;
  if (r >= RR) return;
  float* g = b384 + (size_t)r * 384;
  for (int c = 0; c < HH; ++c)
    g[8 + c] = rbuf[(size_t)r * HH + c] * h[(size_t)r * HH + c];
  for (int c = 72; c < 360; ++c) g[c] = 0.0f;
}

// GRU update: h = u*h + (1-u)*c ; states[0,t,b,n,:] = h
__global__ void update_kernel(const float* __restrict__ ubuf, const float* __restrict__ cbuf,
                              float* __restrict__ h, float* __restrict__ states_out, int t) {
  int i = blockIdx.x * blockDim.x + threadIdx.x;
  if (i >= RR * HH) return;
  int r = i / HH, c = i - r * HH;
  int b = r / NN, n = r - b * NN;
  float u = ubuf[i];
  float hn = u * h[i] + (1.0f - u) * cbuf[i];
  h[i] = hn;
  states_out[((size_t)(t * BB + b) * NN + n) * HH + c] = hn;
}

// ---------------- launcher ----------------
extern "C" void kernel_launch(void* const* d_in, const int* in_sizes, int n_in,
                              void* d_out, int out_size, void* d_ws, size_t ws_size,
                              hipStream_t stream) {
  (void)in_sizes; (void)n_in; (void)out_size; (void)ws_size;
  const float* x    = (const float*)d_in[0];
  const float* mask = (const float*)d_in[1];
  const int*   ei   = (const int*)d_in[2];   // [2,E] int32 (JAX demotes int64)
  const float* ew   = (const float*)d_in[3];
  const float* h0   = (const float*)d_in[4];
  const float* Wfs  = (const float*)d_in[5];
  const float* bfs  = (const float*)d_in[6];
  const float* Wr   = (const float*)d_in[7];
  const float* br   = (const float*)d_in[8];
  const float* Wu   = (const float*)d_in[9];
  const float* bu   = (const float*)d_in[10];
  const float* Wc   = (const float*)d_in[11];
  const float* bc   = (const float*)d_in[12];
  const float* Win  = (const float*)d_in[13];
  const float* bin  = (const float*)d_in[14];
  const float* Wgc  = (const float*)d_in[15];
  const float* bgc  = (const float*)d_in[16];
  const float* Wout = (const float*)d_in[17];
  const float* bout = (const float*)d_in[18];
  const float* Wro  = (const float*)d_in[19];
  const float* bro  = (const float*)d_in[20];
  const float* alphap = (const float*)d_in[21];
  const int* src = ei;
  const int* dst = ei + EE;

  // outputs: imputations[B,T,N,4] | predictions[B,T,N,4] | representations[B,T,N,128] | states[1,T,B,N,64]
  float* imp    = (float*)d_out;
  float* pred   = imp  + (size_t)BB * TT * NN * 4;
  float* reps   = pred + (size_t)BB * TT * NN * 4;
  float* states = reps + (size_t)BB * TT * NN * 128;

  // workspace carve-up (~115 MB fp32; fits 192MB global L2 -> intermediates stay on-chip)
  float* ws = (float*)d_ws;
  size_t off = 0;
  auto nxt = [&](size_t cnt) { float* p = ws + off; off += cnt; return p; };
  float* degf  = nxt(NN);
  float* degb  = nxt(NN);
  float* wf    = nxt(EE);
  float* wb    = nxt(EE);
  float* h     = nxt((size_t)RR * HH);
  float* buf96 = nxt((size_t)RR * 96);   // [x1|m|h|0pad]         (K=72 -> KPAD=96)
  float* zbuf  = nxt((size_t)RR * 64);
  float* b128a = nxt((size_t)RR * 128);  // [Az | A^T z]
  float* b128b = nxt((size_t)RR * 128);  // [gc | h]
  float* b128c = nxt((size_t)RR * 128);  // rep = [o | h]
  float* b384  = nxt((size_t)RR * 384);  // [x|Ax|A2x|ATx|AT2x|0pad] (K=360 -> KPAD=384)
  float* rbuf  = nxt((size_t)RR * HH);
  float* ubuf  = nxt((size_t)RR * HH);
  float* cbuf  = nxt((size_t)RR * HH);
  // packed f16 weight fragments (allocated in float units; 2 halfs per float)
  _Float16* Winp  = (_Float16*)nxt((size_t)96 * 64 / 2);
  _Float16* Wgcp  = (_Float16*)nxt((size_t)128 * 64 / 2);
  _Float16* Woutp = (_Float16*)nxt((size_t)128 * 64 / 2);
  _Float16* Wrp   = (_Float16*)nxt((size_t)384 * 64 / 2);
  _Float16* Wup   = (_Float16*)nxt((size_t)384 * 64 / 2);
  _Float16* Wcp   = (_Float16*)nxt((size_t)384 * 64 / 2);

  const int TPB = 256;
  dim3 mmGrid(RR / 16), mmBlock(128);  // 4 wave32 per block, one per 16-col tile

  // one-time (per launch, deterministic): norms, h0 broadcast, K-padding zeroes, weight packs
  zero_f32_kernel<<<(2 * NN + TPB - 1) / TPB, TPB, 0, stream>>>(degf, 2 * NN);
  deg_accum_kernel<<<(EE + TPB - 1) / TPB, TPB, 0, stream>>>(src, dst, ew, degf, degb);
  norm_w_kernel<<<(EE + TPB - 1) / TPB, TPB, 0, stream>>>(src, dst, ew, degf, degb, wf, wb);
  init_h_kernel<<<(RR * HH + TPB - 1) / TPB, TPB, 0, stream>>>(h0, h);
  zero_f32_kernel<<<(RR * 96 + TPB - 1) / TPB, TPB, 0, stream>>>(buf96, RR * 96);
  zero_f32_kernel<<<((int)((size_t)RR * 384 + TPB - 1) / TPB, TPB), TPB, 0, stream>>>(b384, RR * 384);
  pack_b_kernel<<<(3 * 128 + TPB - 1) / TPB, TPB, 0, stream>>>(Win, 72, 96, Winp);
  pack_b_kernel<<<(4 * 128 + TPB - 1) / TPB, TPB, 0, stream>>>(Wgc, 128, 128, Wgcp);
  pack_b_kernel<<<(4 * 128 + TPB - 1) / TPB, TPB, 0, stream>>>(Wout, 128, 128, Woutp);
  pack_b_kernel<<<(12 * 128 + TPB - 1) / TPB, TPB, 0, stream>>>(Wr, 360, 384, Wrp);
  pack_b_kernel<<<(12 * 128 + TPB - 1) / TPB, TPB, 0, stream>>>(Wu, 360, 384, Wup);
  pack_b_kernel<<<(12 * 128 + TPB - 1) / TPB, TPB, 0, stream>>>(Wc, 360, 384, Wcp);

  for (int t = 0; t < TT; ++t) {
    // stage 1 + concat builds + scatter-target zeroing
    pre_kernel<<<(RR + TPB - 1) / TPB, TPB, 0, stream>>>(
        x, mask, h, Wfs, bfs, pred, buf96, b128a, b128b, b128c, t);

    // z = [x1|m|h] @ W_in + b_in
    wmma_mm_kernel<96><<<mmGrid, mmBlock, 0, stream>>>(buf96, Winp, bin, zbuf, 64, 0, nullptr);

    // gc diffusion (k=1, both directions)
    prop_dir_kernel<<<(EE * 64 + TPB - 1) / TPB, TPB, 0, stream>>>(zbuf, 64, b128a,      128, 64, src, dst, wf);
    prop_dir_kernel<<<(EE * 64 + TPB - 1) / TPB, TPB, 0, stream>>>(zbuf, 64, b128a + 64, 128, 64, dst, src, wb);

    // gc = [Az|ATz] @ W_gc + b ; o = PReLU([gc|h] @ W_out + b)
    wmma_mm_kernel<128><<<mmGrid, mmBlock, 0, stream>>>(b128a, Wgcp,  bgc,  b128b, 128, 0, nullptr);
    wmma_mm_kernel<128><<<mmGrid, mmBlock, 0, stream>>>(b128b, Woutp, bout, b128c, 128, 3, alphap);

    // xhat2 / x2 / outputs, build xg = [x2|m|h] into b384
    mid_kernel<<<(RR + TPB - 1) / TPB, TPB, 0, stream>>>(
        mask, buf96, b128c, Wro, bro, imp, reps, b384, t);

    // 2-hop diffusion of xg (fwd + bwd)
    prop_dir_kernel<<<(EE * 72 + TPB - 1) / TPB, TPB, 0, stream>>>(b384,       384, b384 + 72,  384, 72, src, dst, wf);
    prop_dir_kernel<<<(EE * 72 + TPB - 1) / TPB, TPB, 0, stream>>>(b384,       384, b384 + 216, 384, 72, dst, src, wb);
    prop_dir_kernel<<<(EE * 72 + TPB - 1) / TPB, TPB, 0, stream>>>(b384 + 72,  384, b384 + 144, 384, 72, src, dst, wf);
    prop_dir_kernel<<<(EE * 72 + TPB - 1) / TPB, TPB, 0, stream>>>(b384 + 216, 384, b384 + 288, 384, 72, dst, src, wb);

    // r, u gates (K = 360 padded to 384)
    wmma_mm_kernel<384><<<mmGrid, mmBlock, 0, stream>>>(b384, Wrp, br, rbuf, 64, 1, nullptr);
    wmma_mm_kernel<384><<<mmGrid, mmBlock, 0, stream>>>(b384, Wup, bu, ubuf, 64, 1, nullptr);

    // candidate input: [xi | r*h], re-diffuse
    candprep_kernel<<<(RR + TPB - 1) / TPB, TPB, 0, stream>>>(rbuf, h, b384);
    prop_dir_kernel<<<(EE * 72 + TPB - 1) / TPB, TPB, 0, stream>>>(b384,       384, b384 + 72,  384, 72, src, dst, wf);
    prop_dir_kernel<<<(EE * 72 + TPB - 1) / TPB, TPB, 0, stream>>>(b384,       384, b384 + 216, 384, 72, dst, src, wb);
    prop_dir_kernel<<<(EE * 72 + TPB - 1) / TPB, TPB, 0, stream>>>(b384 + 72,  384, b384 + 144, 384, 72, src, dst, wf);
    prop_dir_kernel<<<(EE * 72 + TPB - 1) / TPB, TPB, 0, stream>>>(b384 + 216, 384, b384 + 288, 384, 72, dst, src, wb);

    // c = tanh(... @ W_c + b) ; h = u*h + (1-u)*c ; states
    wmma_mm_kernel<384><<<mmGrid, mmBlock, 0, stream>>>(b384, Wcp, bc, cbuf, 64, 2, nullptr);
    update_kernel<<<(RR * HH + TPB - 1) / TPB, TPB, 0, stream>>>(ubuf, cbuf, h, states, t);
  }
}
